// RNNModel_3272765079695
// MI455X (gfx1250) — matrix operand order repaired and checked by reading, not compiled, not run on hardware
//
#include <hip/hip_runtime.h>

// ---------------------------------------------------------------------------
// 2-layer LSTM (B=128, T=512, IN=H=512) + FC head, for gfx1250 (MI455X).
// Matrix work on v_wmma_f32_16x16x32_bf16; gate math in fp32.
// One fused GEMM+pointwise kernel per (layer, timestep); weights packed as
// bf16 [4H, IN+H] so each step is gates = bias + [x_t | h_prev] @ Wcat^T.
// A-matrix ([x_t | h_prev] rows for this block) is staged in LDS via
// GLOBAL_LOAD_ASYNC_TO_LDS_B128 (ASYNCcnt) and consumed with ds_load_b128;
// B (weights, 4 MB/layer) streams from global and stays L2-resident (192 MB).
// ---------------------------------------------------------------------------

typedef __attribute__((ext_vector_type(16))) __bf16 v16bf;
typedef __attribute__((ext_vector_type(8)))  float  v8f;

#define B_    128
#define T_    512
#define IN_   512
#define H_    512
#define G4_   2048   // 4*H
#define KCAT_ 1024   // IN + H
#define OUT_  2

// ---- helpers --------------------------------------------------------------

__device__ __forceinline__ unsigned short f32_bf16(float f) {
  unsigned int u = __float_as_uint(f);
  u += 0x7FFFu + ((u >> 16) & 1u);          // round-to-nearest-even
  return (unsigned short)(u >> 16);
}

__device__ __forceinline__ float sigm(float x) {
  return 1.0f / (1.0f + __expf(-x));
}
__device__ __forceinline__ float tanh_fast(float x) {
  return 1.0f - 2.0f / (__expf(2.0f * x) + 1.0f);
}

union Frag32B { uint4 u[2]; v16bf v; };

// B fragment (32x16 bf16, KxN): lane L holds column N=L%16; half selects
// K-range [K0 + half*16, K0 + half*16 + 15] -> 32 contiguous bytes.
__device__ __forceinline__ v16bf load_b(const unsigned short* colPtr, int K0) {
  Frag32B f;
  const uint4* p = (const uint4*)(colPtr + K0);
  f.u[0] = p[0];
  f.u[1] = p[1];
  return f.v;
}

// ---- prep kernels ---------------------------------------------------------

__global__ void cvt_x_bf16(const float* __restrict__ x,
                           unsigned short* __restrict__ xb, int n4) {
  int i = blockIdx.x * blockDim.x + threadIdx.x;
  if (i < n4) {
    float4 f = ((const float4*)x)[i];
    ushort4 o;
    o.x = f32_bf16(f.x); o.y = f32_bf16(f.y);
    o.z = f32_bf16(f.z); o.w = f32_bf16(f.w);
    ((ushort4*)xb)[i] = o;
  }
}

// Wcat[n][k] = (k < IN) ? W_ih[n][k] : W_hh[n][k-IN], as bf16. n in [0,4H)
__global__ void pack_wcat(const float* __restrict__ Wih,
                          const float* __restrict__ Whh,
                          unsigned short* __restrict__ Wcat) {
  int i = blockIdx.x * blockDim.x + threadIdx.x;   // 0 .. 4H*KCAT
  int n = i >> 10;          // KCAT_ = 1024
  int k = i & (KCAT_ - 1);
  float v = (k < IN_) ? Wih[n * IN_ + k] : Whh[n * H_ + (k - IN_)];
  Wcat[i] = f32_bf16(v);
}

__global__ void make_bias(const float* __restrict__ bih,
                          const float* __restrict__ bhh,
                          float* __restrict__ bias) {
  int i = blockIdx.x * blockDim.x + threadIdx.x;
  if (i < G4_) bias[i] = bih[i] + bhh[i];
}

__global__ void init_state(unsigned short* h0, unsigned short* h1,
                           float* c0, float* c1) {
  int i = blockIdx.x * blockDim.x + threadIdx.x;   // B*H = 65536
  h0[i] = 0; h1[i] = 0; c0[i] = 0.0f; c1[i] = 0.0f;
}

// ---- fused LSTM step ------------------------------------------------------
// grid = 32 blocks x 256 threads (8 waves). All waves of a block share the
// same 16-row batch block rb = blockIdx/4; wave picks cb = (blockIdx%4)*8+wave.
// Block stages A = [x_t | h_prev] (16 x 1024 bf16 = 32 KB) in LDS with async
// global->LDS copies, then each wave computes the 4 gate tiles (i,f,g,o) for
// its (rb,cb) with K=1024 and applies the LSTM pointwise update in-register.
__global__ __launch_bounds__(256, 1)
void lstm_step(const unsigned short* __restrict__ xbase, int xstride,
               const unsigned short* __restrict__ hprev,
               const unsigned short* __restrict__ Wcat,
               const float* __restrict__ bias,
               float* __restrict__ c,
               unsigned short* __restrict__ hnext,
               float* __restrict__ hout) {
  __shared__ unsigned short As[16 * KCAT_];   // 32 KB: 16 rows x 1024 K

  const int tid  = threadIdx.x;
  const int lane = tid & 31;
  const int wave = tid >> 5;
  const int rb   = blockIdx.x >> 2;                  // batch block (uniform)
  const int cb   = ((blockIdx.x & 3) << 3) | wave;   // unit block
  const int half = lane >> 4;
  const int ln   = lane & 15;

  // ---- async stage A into LDS: 2048 x 16-byte chunks, 8 per thread -------
  // chunk c: row = c>>7; sub = c&127; sub<64 -> x_t half, else h_prev half.
  {
    const unsigned int ldsBase = (unsigned int)(size_t)&As[0];
    for (int ch = tid; ch < 2048; ch += 256) {
      const int row = ch >> 7;
      const int sub = ch & 127;
      const unsigned short* g;
      unsigned int loff;
      if (sub < 64) {
        g    = xbase + (size_t)(rb * 16 + row) * (size_t)xstride + sub * 8;
        loff = ldsBase + (unsigned int)(row * KCAT_ + sub * 8) * 2u;
      } else {
        g    = hprev + (size_t)(rb * 16 + row) * H_ + (sub - 64) * 8;
        loff = ldsBase + (unsigned int)(row * KCAT_ + IN_ + (sub - 64) * 8) * 2u;
      }
      asm volatile("global_load_async_to_lds_b128 %0, %1, off"
                   :: "v"(loff), "v"(g) : "memory");
    }
    asm volatile("s_wait_asynccnt 0x0" ::: "memory");
  }
  __syncthreads();

  // ---- B column pointers: row n of Wcat is contiguous in k ----------------
  const int nco = cb * 16 + ln;
  const unsigned short* b0 = Wcat + (size_t)(0 * H_ + nco) * KCAT_ + half * 16;
  const unsigned short* b1 = Wcat + (size_t)(1 * H_ + nco) * KCAT_ + half * 16;
  const unsigned short* b2 = Wcat + (size_t)(2 * H_ + nco) * KCAT_ + half * 16;
  const unsigned short* b3 = Wcat + (size_t)(3 * H_ + nco) * KCAT_ + half * 16;

  v8f acc0 = {0,0,0,0,0,0,0,0};
  v8f acc1 = {0,0,0,0,0,0,0,0};
  v8f acc2 = {0,0,0,0,0,0,0,0};
  v8f acc3 = {0,0,0,0,0,0,0,0};

  // A fragment (16x32 bf16, MxK) from LDS: lane holds row M=ln;
  // elements 0..15 = K0 + half*16 .. K0 + half*16 + 15.
  const unsigned short* aRow = &As[ln * KCAT_];

#pragma unroll
  for (int kk = 0; kk < 32; ++kk) {
    const int K0 = kk * 32;
    Frag32B a;
    a.u[0] = *(const uint4*)(aRow + K0 + half * 16);
    a.u[1] = *(const uint4*)(aRow + K0 + half * 16 + 8);
    acc0 = __builtin_amdgcn_wmma_f32_16x16x32_bf16(false, a.v, false, load_b(b0, K0), (short)0, acc0, false, false);
    acc1 = __builtin_amdgcn_wmma_f32_16x16x32_bf16(false, a.v, false, load_b(b1, K0), (short)0, acc1, false, false);
    acc2 = __builtin_amdgcn_wmma_f32_16x16x32_bf16(false, a.v, false, load_b(b2, K0), (short)0, acc2, false, false);
    acc3 = __builtin_amdgcn_wmma_f32_16x16x32_bf16(false, a.v, false, load_b(b3, K0), (short)0, acc3, false, false);
  }

  // ---- pointwise LSTM update. C/D tile layout: element v of the v8f lives
  // at (m = v + 8*half, n = ln) of the 16x16 tile. ---------------------------
  const int j = cb * 16 + ln;
#pragma unroll
  for (int v = 0; v < 8; ++v) {
    const int brow = rb * 16 + v + half * 8;
    const size_t idx = (size_t)brow * H_ + j;
    float gi = sigm(acc0[v] + bias[j]);
    float gf = sigm(acc1[v] + bias[H_ + j]);
    float gg = tanh_fast(acc2[v] + bias[2 * H_ + j]);
    float go = sigm(acc3[v] + bias[3 * H_ + j]);
    float cn = gf * c[idx] + gi * gg;
    c[idx] = cn;
    float hn = go * tanh_fast(cn);
    hnext[idx] = f32_bf16(hn);
    if (hout) hout[idx] = hn;
  }
}

// ---- FC head --------------------------------------------------------------

__global__ void fc_kernel(const float* __restrict__ h,
                          const float* __restrict__ w,
                          const float* __restrict__ b,
                          float* __restrict__ out) {
  int i = threadIdx.x;            // 256 threads = 128 batches x 2 outputs
  int bb = i >> 1, o = i & 1;
  const float* hr = h + (size_t)bb * H_;
  const float* wr = w + (size_t)o * H_;
  float s = 0.0f;
  for (int k = 0; k < H_; ++k) s += hr[k] * wr[k];
  out[bb * OUT_ + o] = s + b[o];
}

// ---- launch ---------------------------------------------------------------

extern "C" void kernel_launch(void* const* d_in, const int* in_sizes, int n_in,
                              void* d_out, int out_size, void* d_ws, size_t ws_size,
                              hipStream_t stream) {
  const float* x    = (const float*)d_in[0];
  const float* Wih0 = (const float*)d_in[1];
  const float* Whh0 = (const float*)d_in[2];
  const float* bih0 = (const float*)d_in[3];
  const float* bhh0 = (const float*)d_in[4];
  const float* Wih1 = (const float*)d_in[5];
  const float* Whh1 = (const float*)d_in[6];
  const float* bih1 = (const float*)d_in[7];
  const float* bhh1 = (const float*)d_in[8];
  const float* fcw  = (const float*)d_in[9];
  const float* fcb  = (const float*)d_in[10];
  float* out = (float*)d_out;

  char* ws = (char*)d_ws;
  size_t off = 0;
  auto alloc = [&](size_t bytes) -> char* {
    char* p = ws + off;
    off += (bytes + 255) & ~(size_t)255;
    return p;
  };

  unsigned short* Wcat0 = (unsigned short*)alloc((size_t)G4_ * KCAT_ * 2);
  unsigned short* Wcat1 = (unsigned short*)alloc((size_t)G4_ * KCAT_ * 2);
  float* bias0 = (float*)alloc(G4_ * 4);
  float* bias1 = (float*)alloc(G4_ * 4);
  float* c0    = (float*)alloc((size_t)B_ * H_ * 4);
  float* c1    = (float*)alloc((size_t)B_ * H_ * 4);
  unsigned short* h0buf[2];
  unsigned short* h1buf[2];
  h0buf[0] = (unsigned short*)alloc((size_t)B_ * H_ * 2);
  h0buf[1] = (unsigned short*)alloc((size_t)B_ * H_ * 2);
  h1buf[0] = (unsigned short*)alloc((size_t)B_ * H_ * 2);
  h1buf[1] = (unsigned short*)alloc((size_t)B_ * H_ * 2);
  float* h1f = (float*)alloc((size_t)B_ * H_ * 4);
  unsigned short* xb = (unsigned short*)alloc((size_t)B_ * T_ * IN_ * 2);

  // Prep: pack weights/bias to bf16/fp32, convert x to bf16, zero states.
  pack_wcat<<<(G4_ * KCAT_) / 256, 256, 0, stream>>>(Wih0, Whh0, Wcat0);
  pack_wcat<<<(G4_ * KCAT_) / 256, 256, 0, stream>>>(Wih1, Whh1, Wcat1);
  make_bias<<<8, 256, 0, stream>>>(bih0, bhh0, bias0);
  make_bias<<<8, 256, 0, stream>>>(bih1, bhh1, bias1);
  {
    int n4 = (B_ * T_ * IN_) / 4;
    cvt_x_bf16<<<(n4 + 255) / 256, 256, 0, stream>>>(x, xb, n4);
  }
  init_state<<<(B_ * H_) / 256, 256, 0, stream>>>(h0buf[0], h1buf[0], c0, c1);

  // Sequential scan: per timestep, layer0 then layer1 (fused GEMM+pointwise).
  const int xstride0 = T_ * IN_;   // batch-row stride in x
  for (int t = 0; t < T_; ++t) {
    const unsigned short* h0r = h0buf[t & 1];
    unsigned short*       h0w = h0buf[(t + 1) & 1];
    lstm_step<<<32, 256, 0, stream>>>(xb + (size_t)t * IN_, xstride0,
                                      h0r, Wcat0, bias0, c0, h0w, nullptr);
    const unsigned short* h1r = h1buf[t & 1];
    unsigned short*       h1w = h1buf[(t + 1) & 1];
    lstm_step<<<32, 256, 0, stream>>>(h0w, H_,
                                      h1r, Wcat1, bias1, c1, h1w,
                                      (t == T_ - 1) ? h1f : nullptr);
  }

  fc_kernel<<<1, 256, 0, stream>>>(h1f, fcw, fcb, out);
}
